// TransformerModel_45251775431036
// MI455X (gfx1250) — compile-verified, matrix operand order checked by reading, not run on hardware
//
#include <hip/hip_runtime.h>
#include <hip/hip_bf16.h>

// Problem constants (match reference)
#define VOCAB 32000
#define CDIM  256
#define BB    4
#define TT    2048
#define MM    (BB*TT)      // 8192 rows total

typedef __attribute__((ext_vector_type(16))) __bf16 v16bf;
typedef __attribute__((ext_vector_type(8)))  float  v8f;

// ---------- helpers ----------
__device__ __forceinline__ unsigned short f2bf(float f) {
    unsigned int u = __float_as_uint(f);
    u += 0x7FFFu + ((u >> 16) & 1u);      // round-to-nearest-even
    return (unsigned short)(u >> 16);
}
__device__ __forceinline__ __bf16 bfraw(unsigned short u) {
    union { unsigned short u; __bf16 b; } c; c.u = u; return c.b;
}
__device__ __forceinline__ v8f zero8() {
    v8f z = {0.f,0.f,0.f,0.f,0.f,0.f,0.f,0.f}; return z;
}

// Load one 16-bit WMMA A/B fragment (16 halves / lane) for a 32-wide K-chunk.
// `row` points to the lane's 32 contiguous bf16 K-values; `base` = (lane>=16)?8:0.
// Halves 0..7  <- K = base..base+7       (VGPRs 0..3)
// Halves 8..15 <- K = 16+base..16+base+7 (VGPRs 4..7)
struct U32x8 { unsigned int w[8]; };
__device__ __forceinline__ v16bf frag_load(const unsigned short* row, int base) {
    U32x8 t;
    const unsigned int* plo = reinterpret_cast<const unsigned int*>(row + base);
    const unsigned int* phi = reinterpret_cast<const unsigned int*>(row + 16 + base);
#pragma unroll
    for (int i = 0; i < 4; ++i) { t.w[i] = plo[i]; t.w[4+i] = phi[i]; }
    return __builtin_bit_cast(v16bf, t);
}

__device__ __forceinline__ v8f wmma_bf16(v16bf a, v16bf b, v8f c) {
    return __builtin_amdgcn_wmma_f32_16x16x32_bf16(
        /*neg_a=*/false, a, /*neg_b=*/false, b,
        /*c_mod=*/(short)0, c, /*reuse_a=*/false, /*reuse_b=*/false);
}

// ---------- kernel 1: h = tok_emb[x] + pos_emb  (fp32 -> bf16) ----------
__global__ __launch_bounds__(256)
void k_embed(const int* __restrict__ x, const float* __restrict__ tok,
             const float* __restrict__ pos, unsigned short* __restrict__ hb) {
    int i = blockIdx.x * 256 + threadIdx.x;    // over MM*CDIM
    int c = i & (CDIM - 1);
    int m = i >> 8;
    int t = m & (TT - 1);
    float v = tok[(size_t)x[m]*CDIM + c] + pos[(size_t)t*CDIM + c];
    hb[i] = f2bf(v);
}

// ---------- kernel 2: transpose+convert Wq/Wk/Wv to bf16 [Cout][Cin] ----------
__global__ __launch_bounds__(256)
void k_wqkv_t(const float* __restrict__ Wq, const float* __restrict__ Wk,
              const float* __restrict__ Wv, unsigned short* __restrict__ Wqt,
              unsigned short* __restrict__ Wkt, unsigned short* __restrict__ Wvt) {
    int i = blockIdx.x * 256 + threadIdx.x;    // over 3*CDIM*CDIM
    int mat = i / (CDIM*CDIM);
    int r   = i % (CDIM*CDIM);
    int d = r >> 8, c = r & (CDIM - 1);        // Wt[d][c] = W[c][d]
    const float* W = (mat == 0) ? Wq : (mat == 1) ? Wk : Wv;
    unsigned short* Wt = (mat == 0) ? Wqt : (mat == 1) ? Wkt : Wvt;
    Wt[r] = f2bf(W[(size_t)c*CDIM + d]);
}

// ---------- kernel 3: transpose+convert Wp to bf16 [VOCAB][CDIM] ----------
__global__ __launch_bounds__(256)
void k_wp_t(const float* __restrict__ Wp, unsigned short* __restrict__ Wpt) {
    int i = blockIdx.x * 256 + threadIdx.x;    // over VOCAB*CDIM
    int c = i & (CDIM - 1);
    int v = i >> 8;
    Wpt[i] = f2bf(Wp[(size_t)c*VOCAB + v]);
}

// ---------- kernel 4: QKV GEMM  Y = h @ W + b  (WMMA bf16) ----------
// grid (MM/16, CDIM/16, 3), 1 wave/block. Q pre-scaled by 1/sqrt(C)=1/16.
// V is stored transposed: Vt[b][c][t].
__global__ __launch_bounds__(32)
void k_qkv(const unsigned short* __restrict__ hb,
           const unsigned short* __restrict__ Wqt, const unsigned short* __restrict__ Wkt,
           const unsigned short* __restrict__ Wvt,
           const float* __restrict__ bq, const float* __restrict__ bk,
           const float* __restrict__ bv,
           unsigned short* __restrict__ Qb, unsigned short* __restrict__ Kb,
           unsigned short* __restrict__ Vt) {
    const int l = threadIdx.x;
    const int half = l >> 4, lane16 = l & 15, base = half * 8;
    const int m0 = blockIdx.x * 16;
    const int n0 = blockIdx.y * 16;
    const int mat = blockIdx.z;
    const unsigned short* Wt = (mat == 0) ? Wqt : (mat == 1) ? Wkt : Wvt;
    const float* bias = (mat == 0) ? bq : (mat == 1) ? bk : bv;

    const unsigned short* arow = hb + (size_t)(m0 + lane16) * CDIM;
    const unsigned short* brow = Wt + (size_t)(n0 + lane16) * CDIM;
    v8f acc = zero8();
#pragma unroll
    for (int kc = 0; kc < CDIM; kc += 32) {
        v16bf a = frag_load(arow + kc, base);
        v16bf b = frag_load(brow + kc, base);
        acc = wmma_bf16(a, b, acc);
    }
    const float bsc = bias[n0 + lane16];
#pragma unroll
    for (int i = 0; i < 8; ++i) {
        int m = m0 + i + 8 * half;
        int n = n0 + lane16;
        float v = acc[i] + bsc;
        if (mat == 0) {
            Qb[(size_t)m*CDIM + n] = f2bf(v * 0.0625f);   // fold 1/sqrt(256)
        } else if (mat == 1) {
            Kb[(size_t)m*CDIM + n] = f2bf(v);
        } else {
            int b = m >> 11;              // /TT
            int t = m & (TT - 1);
            Vt[((size_t)b*CDIM + n)*TT + t] = f2bf(v);
        }
    }
}

// ---------- kernel 5: flash-style causal attention, 4 waves / query tile ----
// grid (TT/16, BB), blockDim 128. The d=256 dimension is split across the 4
// waves: wave w computes partial scores over d-slice [64w, 64w+64) (2 WMMAs),
// partials are reduced in LDS, softmax runs on lanes 0..15, and wave w owns
// the 64-column O slice [64w, 64w+64) (4 v8f accumulators).
__global__ __launch_bounds__(128)
void k_attn(const unsigned short* __restrict__ Qb, const unsigned short* __restrict__ Kb,
            const unsigned short* __restrict__ Vt, unsigned short* __restrict__ Ctxb) {
    __shared__ float tilep[4][16][32];   // per-wave partial scores (8 KB)
    __shared__ float tile[16][32];       // masked probabilities  (2 KB)
    __shared__ float salpha[16];
    __shared__ float sm[16];
    __shared__ float sl[16];

    const int tid = threadIdx.x;
    const int w = tid >> 5;              // wave id 0..3
    const int l = tid & 31;
    const int half = l >> 4, lane16 = l & 15, base = half * 8;
    const int b  = blockIdx.y;
    const int m0 = blockIdx.x * 16;
    const int mg0 = b * TT + m0;

    // Q fragments for this wave's d-slice
    v16bf qf0, qf1;
    {
        const unsigned short* qrow = Qb + (size_t)(mg0 + lane16) * CDIM + w * 64;
        qf0 = frag_load(qrow, base);
        qf1 = frag_load(qrow + 32, base);
    }

    v8f o[4];
#pragma unroll
    for (int f = 0; f < 4; ++f) o[f] = zero8();

    if (tid < 16) { sm[tid] = -1e30f; sl[tid] = 0.f; }
    __syncthreads();

    const int send = m0 + 16;            // keys needed: s < send (causal)
    for (int s0 = 0; s0 < send; s0 += 32) {
        // ---- partial scores over this wave's 64-wide d-slice ----
#pragma unroll
        for (int fr = 0; fr < 2; ++fr) {
            int s = s0 + fr * 16 + lane16;
            const unsigned short* krow = Kb + (size_t)(b*TT + s) * CDIM + w * 64;
            v8f sc = zero8();
            sc = wmma_bf16(qf0, frag_load(krow,      base), sc);
            sc = wmma_bf16(qf1, frag_load(krow + 32, base), sc);
#pragma unroll
            for (int i = 0; i < 8; ++i)
                tilep[w][i + 8*half][fr*16 + lane16] = sc[i];
        }
        __syncthreads();

        // ---- reduce partials + online softmax (one lane per query row) ----
        if (tid < 16) {
            int t = m0 + tid;
            float mold = sm[tid];
            float cmax = -1e30f;
#pragma unroll
            for (int j = 0; j < 32; ++j) {
                float v = tilep[0][tid][j] + tilep[1][tid][j]
                        + tilep[2][tid][j] + tilep[3][tid][j];
                v = (s0 + j <= t) ? v : -1e30f;
                tile[tid][j] = v;
                cmax = fmaxf(cmax, v);
            }
            float mnew = fmaxf(mold, cmax);
            float alpha = __expf(mold - mnew);
            float sum = 0.f;
#pragma unroll
            for (int j = 0; j < 32; ++j) {
                float p = (s0 + j <= t) ? __expf(tile[tid][j] - mnew) : 0.f;
                tile[tid][j] = p;
                sum += p;
            }
            sm[tid] = mnew;
            sl[tid] = alpha * sl[tid] + sum;
            salpha[tid] = alpha;
        }
        __syncthreads();

        // ---- build P fragment (identical in every wave) ----
        v16bf pf;
#pragma unroll
        for (int e = 0; e < 8; ++e) {
            pf[e]     = bfraw(f2bf(tile[lane16][base + e]));
            pf[e + 8] = bfraw(f2bf(tile[lane16][16 + base + e]));
        }
        float al[8];
#pragma unroll
        for (int i = 0; i < 8; ++i) al[i] = salpha[i + 8*half];

        // ---- O_slice = diag(alpha)*O_slice + P @ V_chunk (4 WMMAs) ----
#pragma unroll
        for (int f = 0; f < 4; ++f) {
#pragma unroll
            for (int i = 0; i < 8; ++i) o[f][i] = o[f][i] * al[i];
            const unsigned short* vrow =
                Vt + ((size_t)b*CDIM + w*64 + f*16 + lane16) * TT + s0;
            o[f] = wmma_bf16(pf, frag_load(vrow, base), o[f]);
        }
        __syncthreads();   // protect tile/tilep/salpha for next chunk
    }

    // ---- normalize and store this wave's ctx slice (bf16) ----
#pragma unroll
    for (int f = 0; f < 4; ++f) {
#pragma unroll
        for (int i = 0; i < 8; ++i) {
            int r = i + 8*half;
            float v = o[f][i] / sl[r];
            Ctxb[(size_t)(mg0 + r)*CDIM + w*64 + f*16 + lane16] = f2bf(v);
        }
    }
}

// ---------- kernel 6: logits = ctx @ Wp + bp  (WMMA bf16, NT stores) --------
// grid (MM/32, VOCAB/64), 1 wave/block computes a 32x64 tile: two A-row sets
// share every B fragment (halves Wpt L2 traffic vs 16-row tiles), 64 WMMAs.
__global__ __launch_bounds__(32)
void k_logits(const unsigned short* __restrict__ Ctxb, const unsigned short* __restrict__ Wpt,
              const float* __restrict__ bp, float* __restrict__ out) {
    const int l = threadIdx.x;
    const int half = l >> 4, lane16 = l & 15, base = half * 8;
    const int m0  = blockIdx.x * 32;
    const int n00 = blockIdx.y * 64;

    v16bf af[2][8];
#pragma unroll
    for (int r = 0; r < 2; ++r) {
        const unsigned short* arow = Ctxb + (size_t)(m0 + r*16 + lane16) * CDIM;
#pragma unroll
        for (int kc = 0; kc < 8; ++kc) af[r][kc] = frag_load(arow + kc * 32, base);
    }

#pragma unroll
    for (int nt = 0; nt < 4; ++nt) {
        int n0 = n00 + nt * 16;
        const unsigned short* brow = Wpt + (size_t)(n0 + lane16) * CDIM;
        v8f acc0 = zero8(), acc1 = zero8();
#pragma unroll
        for (int kc = 0; kc < 8; ++kc) {
            v16bf bf = frag_load(brow + kc * 32, base);
            acc0 = wmma_bf16(af[0][kc], bf, acc0);   // independent chains ->
            acc1 = wmma_bf16(af[1][kc], bf, acc1);   // back-to-back XDL issue
        }
        float bias = bp[n0 + lane16];
#pragma unroll
        for (int i = 0; i < 8; ++i) {
            int mA = m0 + i + 8*half;
            int mB = m0 + 16 + i + 8*half;
            // 1 GB stream-out: non-temporal so Wpt stays resident in L2
            __builtin_nontemporal_store(acc0[i] + bias,
                                        out + (size_t)mA*VOCAB + n0 + lane16);
            __builtin_nontemporal_store(acc1[i] + bias,
                                        out + (size_t)mB*VOCAB + n0 + lane16);
        }
    }
}

// ---------- kernel 7: per-row cross-entropy, single-pass online LSE --------
__global__ __launch_bounds__(256)
void k_rowloss(const float* __restrict__ logits, const int* __restrict__ tgt,
               float* __restrict__ rowloss) {
    __shared__ float rm[256];
    __shared__ float rs[256];
    const int r = blockIdx.x;
    const int t = threadIdx.x;
    const float* row = logits + (size_t)r * VOCAB;

    // one streaming pass: running (max, scaled-sum) pair per thread
    float m = -1e30f, s = 0.f;
    for (int j = t; j < VOCAB; j += 256) {
        float v = __builtin_nontemporal_load(row + j);
        float mn = fmaxf(m, v);
        s = s * __expf(m - mn) + __expf(v - mn);   // exp on TRANS pipe
        m = mn;
    }
    rm[t] = m; rs[t] = s; __syncthreads();
    for (int k = 128; k > 0; k >>= 1) {
        if (t < k) {
            float m2 = rm[t+k], s2 = rs[t+k];
            float mn = fmaxf(rm[t], m2);
            rs[t] = rs[t] * __expf(rm[t] - mn) + s2 * __expf(m2 - mn);
            rm[t] = mn;
        }
        __syncthreads();
    }
    if (t == 0) rowloss[r] = (rm[0] + __logf(rs[0])) - row[tgt[r]];
}

// ---------- kernel 8: mean over rows -> loss scalar ----------
__global__ __launch_bounds__(256)
void k_loss(const float* __restrict__ rowloss, float* __restrict__ out) {
    __shared__ float red[256];
    const int t = threadIdx.x;
    float s = 0.f;
    for (int j = t; j < MM; j += 256) s += rowloss[j];
    red[t] = s; __syncthreads();
    for (int k = 128; k > 0; k >>= 1) { if (t < k) red[t] += red[t+k]; __syncthreads(); }
    if (t == 0) out[(size_t)MM * VOCAB] = red[0] / (float)MM;
}

// ---------- launch ----------
extern "C" void kernel_launch(void* const* d_in, const int* in_sizes, int n_in,
                              void* d_out, int out_size, void* d_ws, size_t ws_size,
                              hipStream_t stream) {
    (void)in_sizes; (void)n_in; (void)out_size; (void)ws_size;
    const int*   x   = (const int*)  d_in[0];
    const int*   tgt = (const int*)  d_in[1];
    const float* tok = (const float*)d_in[2];
    const float* pos = (const float*)d_in[3];
    const float* Wq  = (const float*)d_in[4];
    const float* bq  = (const float*)d_in[5];
    const float* Wk  = (const float*)d_in[6];
    const float* bk  = (const float*)d_in[7];
    const float* Wv  = (const float*)d_in[8];
    const float* bv  = (const float*)d_in[9];
    const float* Wp  = (const float*)d_in[10];
    const float* bp  = (const float*)d_in[11];
    float* out = (float*)d_out;

    // Workspace layout (~37.4 MB total, all bf16 stored as raw ushort)
    unsigned short* hb   = (unsigned short*)d_ws;                 // [MM][CDIM]
    unsigned short* Qb   = hb   + (size_t)MM*CDIM;                // [MM][CDIM] (pre-scaled)
    unsigned short* Kb   = Qb   + (size_t)MM*CDIM;                // [MM][CDIM]
    unsigned short* Vt   = Kb   + (size_t)MM*CDIM;                // [BB][CDIM][TT]
    unsigned short* Ctxb = Vt   + (size_t)BB*CDIM*TT;             // [MM][CDIM]
    unsigned short* Wqt  = Ctxb + (size_t)MM*CDIM;                // [CDIM][CDIM]
    unsigned short* Wkt  = Wqt  + (size_t)CDIM*CDIM;
    unsigned short* Wvt  = Wkt  + (size_t)CDIM*CDIM;
    unsigned short* Wpt  = Wvt  + (size_t)CDIM*CDIM;              // [VOCAB][CDIM]
    float* rowloss = (float*)(Wpt + (size_t)VOCAB*CDIM);          // [MM]

    k_embed <<< (MM*CDIM)/256, 256, 0, stream >>>(x, tok, pos, hb);
    k_wqkv_t<<< (3*CDIM*CDIM)/256, 256, 0, stream >>>(Wq, Wk, Wv, Wqt, Wkt, Wvt);
    k_wp_t  <<< (VOCAB*CDIM)/256, 256, 0, stream >>>(Wp, Wpt);
    k_qkv   <<< dim3(MM/16, CDIM/16, 3), 32, 0, stream >>>(hb, Wqt, Wkt, Wvt,
                                                           bq, bk, bv, Qb, Kb, Vt);
    k_attn  <<< dim3(TT/16, BB), 128, 0, stream >>>(Qb, Kb, Vt, Ctxb);
    k_logits<<< dim3(MM/32, VOCAB/64), 32, 0, stream >>>(Ctxb, Wpt, bp, out);
    k_rowloss<<< MM, 256, 0, stream >>>(out, tgt, rowloss);
    k_loss  <<< 1, 256, 0, stream >>>(rowloss, out);
}
